// EigenbasisSumKernel_83786222010713
// MI455X (gfx1250) — compile-verified, matrix operand order checked by reading, not run on hardware
//
#include <hip/hip_runtime.h>
#include <hip/hip_bf16.h>

typedef __attribute__((ext_vector_type(16))) __bf16 v16bf;
typedef __attribute__((ext_vector_type(8)))  __bf16 v8bf;
typedef __attribute__((ext_vector_type(8)))  float  v8f;

#define N_PTS   4096
#define L_EIG   512
#define K_DIM   1024      // 2*L (cos block | sin block)
#define BLK     128       // C block edge
#define KC      128       // K chunk staged through LDS (4 wmma k-steps)
#define LDS_PAD 8         // 16B row pad: stride 272B = 68 banks == 4 mod 64 -> conflict-free b128
#define LDS_LDK (KC + LDS_PAD)

// ---------------------------------------------------------------------------
// Stage 0: c = |variance| / sum_l exp(-0.5 * lambda_l)
// ---------------------------------------------------------------------------
__global__ __launch_bounds__(256) void eig_scale_kernel(const float* __restrict__ lb,
                                                        const float* __restrict__ var,
                                                        float* __restrict__ cptr) {
    __shared__ float red[256];
    const int t = threadIdx.x;
    float s = __expf(-0.5f * lb[t]) + __expf(-0.5f * lb[t + 256]);
    red[t] = s;
    __syncthreads();
    for (int o = 128; o > 0; o >>= 1) {
        if (t < o) red[t] += red[t + o];
        __syncthreads();
    }
    if (t == 0) cptr[0] = fabsf(var[0]) / red[0];
}

// ---------------------------------------------------------------------------
// Stage 1: pack bf16 operand matrix [n][K_DIM]:
//   cols [0,512)    = cos(<f_l, p_n>) * scale_l
//   cols [512,1024) = sin(<f_l, p_n>) * scale_l
// scale_l = w_l * c for the A side, 1 for the B side.
// ---------------------------------------------------------------------------
__global__ __launch_bounds__(256) void eig_build_kernel(const float* __restrict__ pts,
                                                        const float* __restrict__ freqs,
                                                        const float* __restrict__ lb,
                                                        const float* __restrict__ cptr,
                                                        __bf16* __restrict__ outbuf,
                                                        int applyScale) {
    const int l = blockIdx.x * blockDim.x + threadIdx.x;   // 0..511
    const int n = blockIdx.y;                              // 0..4095
    if (l >= L_EIG) return;
    const float f0 = freqs[l * 3 + 0], f1 = freqs[l * 3 + 1], f2 = freqs[l * 3 + 2];
    const float px = pts[n * 3 + 0] * f0 + pts[n * 3 + 1] * f1 + pts[n * 3 + 2] * f2;
    float sc = 1.0f;
    if (applyScale) sc = __expf(-0.5f * lb[l]) * cptr[0];
    float sv, cv;
    __sincosf(px, &sv, &cv);
    outbuf[(size_t)n * K_DIM + l]         = (__bf16)(cv * sc);
    outbuf[(size_t)n * K_DIM + L_EIG + l] = (__bf16)(sv * sc);
}

// ---------------------------------------------------------------------------
// Fragment load from LDS: 16x32 bf16 tile in the CDNA5 WMMA A/B lane layout.
// Lane (r = lane&15, h = lane>>4):
//   elems 0..7  : K = h*8 + 0..7
//   elems 8..15 : K = 16 + h*8 + 0..7
// Two ds_load_b128 + shuffle.
// ---------------------------------------------------------------------------
__device__ __forceinline__ v16bf load_frag(const __bf16* base, int rowStride,
                                           int row, int k0, int h) {
    const __bf16* p = base + row * rowStride + k0;
    v8bf lo = *(const v8bf*)(p + h * 8);
    v8bf hi = *(const v8bf*)(p + 16 + h * 8);
    return __builtin_shufflevector(lo, hi, 0, 1, 2, 3, 4, 5, 6, 7,
                                   8, 9, 10, 11, 12, 13, 14, 15);
}

// Async global -> LDS copy of 16 bytes (per lane), tracked by ASYNCcnt.
// ldsOff = low 32 bits of the generic address of a __shared__ location
// (flat->LDS mapping truncates to the 32-bit LDS offset, ISA 10.2).
__device__ __forceinline__ void async_copy_b128(unsigned ldsOff, const void* gaddr) {
    asm volatile("global_load_async_to_lds_b128 %0, %1, off"
                 :: "v"(ldsOff), "v"((unsigned long long)(size_t)gaddr)
                 : "memory");
}

__device__ __forceinline__ void wait_async_zero() {
    asm volatile("s_wait_asynccnt 0x0" ::: "memory");
}

// ---------------------------------------------------------------------------
// Stage 2: C[4096x4096] = A[4096xK] * B[4096xK]^T   (K = 1024, bf16 -> f32)
// 256 threads / 8 waves per block; block computes 128x128 of C.
// Wave w: rows (w>>1)*32 .. +31, cols (w&1)*64 .. +63  => 2x4 tiles of 16x16.
// Double-buffered LDS filled by GLOBAL_LOAD_ASYNC_TO_LDS_B128: chunk kc+KC is
// in flight while the 32-WMMA burst runs on chunk kc. One barrier per chunk.
// ---------------------------------------------------------------------------
__global__ __launch_bounds__(256) void eig_gemm_kernel(const __bf16* __restrict__ Ab,
                                                       const __bf16* __restrict__ Bb,
                                                       float* __restrict__ out) {
    __shared__ __bf16 As[2][BLK * LDS_LDK];
    __shared__ __bf16 Bs[2][BLK * LDS_LDK];

    const int t    = threadIdx.x;
    const int lane = t & 31;
    const int wave = t >> 5;          // 0..7
    const int wrow = wave >> 1;       // 0..3 -> 32-row strip
    const int wcol = wave & 1;        // 0..1 -> 64-col strip
    const int r    = lane & 15;
    const int h    = lane >> 4;

    const int blockM = blockIdx.y * BLK;
    const int blockN = blockIdx.x * BLK;

    const v8f vzero = {0.f, 0.f, 0.f, 0.f, 0.f, 0.f, 0.f, 0.f};
    v8f acc[2][4];
    for (int mi = 0; mi < 2; ++mi)
        for (int ni = 0; ni < 4; ++ni)
            acc[mi][ni] = vzero;

    // Global->LDS feed pattern: 16 threads per 256B row, 8 rows per thread.
    const int fr  = t >> 4;             // 0..15
    const int seg = (t & 15) * 8;       // bf16 element offset (16B chunks)

    auto issue_chunk = [&](int kc, int buf) {
        #pragma unroll
        for (int i = 0; i < 8; ++i) {
            const int row = fr + i * 16;
            async_copy_b128((unsigned)(size_t)&As[buf][row * LDS_LDK + seg],
                            Ab + (size_t)(blockM + row) * K_DIM + kc + seg);
            async_copy_b128((unsigned)(size_t)&Bs[buf][row * LDS_LDK + seg],
                            Bb + (size_t)(blockN + row) * K_DIM + kc + seg);
        }
    };

    issue_chunk(0, 0);
    int buf = 0;

    for (int kc = 0; kc < K_DIM; kc += KC) {
        // My async fills of As[buf]/Bs[buf] are done; barrier makes it true
        // for every wave AND proves everyone finished reading buf^1 so it can
        // be overwritten.
        wait_async_zero();
        __syncthreads();

        // Kick off the next chunk into the other buffer; it completes while
        // the WMMA burst below runs.
        if (kc + KC < K_DIM) issue_chunk(kc + KC, buf ^ 1);

        const __bf16* as = As[buf];
        const __bf16* bs = Bs[buf];
        #pragma unroll
        for (int ks = 0; ks < KC; ks += 32) {
            v16bf afrag[2], bfrag[4];
            #pragma unroll
            for (int mi = 0; mi < 2; ++mi)
                afrag[mi] = load_frag(as, LDS_LDK, wrow * 32 + mi * 16 + r, ks, h);
            #pragma unroll
            for (int ni = 0; ni < 4; ++ni)
                bfrag[ni] = load_frag(bs, LDS_LDK, wcol * 64 + ni * 16 + r, ks, h);

            #pragma unroll
            for (int mi = 0; mi < 2; ++mi)
                #pragma unroll
                for (int ni = 0; ni < 4; ++ni)
                    acc[mi][ni] = __builtin_amdgcn_wmma_f32_16x16x32_bf16(
                        false, afrag[mi], false, bfrag[ni],
                        (short)0, acc[mi][ni], false, false);
        }
        buf ^= 1;
    }

    // Epilogue: f32 C tile layout -> VGPR i holds M=i (lanes 0-15), M=8+i (lanes 16-31)
    #pragma unroll
    for (int mi = 0; mi < 2; ++mi) {
        #pragma unroll
        for (int ni = 0; ni < 4; ++ni) {
            const int col = blockN + wcol * 64 + ni * 16 + r;
            #pragma unroll
            for (int i = 0; i < 8; ++i) {
                const int row = blockM + wrow * 32 + mi * 16 + i + h * 8;
                out[(size_t)row * N_PTS + col] = acc[mi][ni][i];
            }
        }
    }
}

extern "C" void kernel_launch(void* const* d_in, const int* in_sizes, int n_in,
                              void* d_out, int out_size, void* d_ws, size_t ws_size,
                              hipStream_t stream) {
    const float* x     = (const float*)d_in[0];   // [4096,3]
    const float* y     = (const float*)d_in[1];   // [4096,3]
    const float* freqs = (const float*)d_in[2];   // [512,3]
    const float* lb    = (const float*)d_in[3];   // [512]
    const float* var   = (const float*)d_in[4];   // [1]
    float* out = (float*)d_out;                   // [4096,4096]

    float*  cptr = (float*)d_ws;
    __bf16* Abuf = (__bf16*)((char*)d_ws + 256);
    __bf16* Bbuf = Abuf + (size_t)N_PTS * K_DIM;

    // Stage 0: global scale c = |var| / sum(w)
    eig_scale_kernel<<<1, 256, 0, stream>>>(lb, var, cptr);

    // Stage 1: pack bf16 operands (scale folded into A side)
    dim3 bgrid(2, N_PTS);
    eig_build_kernel<<<bgrid, 256, 0, stream>>>(x, freqs, lb, cptr, Abuf, 1);
    eig_build_kernel<<<bgrid, 256, 0, stream>>>(y, freqs, lb, cptr, Bbuf, 0);

    // Stage 2: WMMA GEMM
    dim3 ggrid(N_PTS / BLK, N_PTS / BLK);
    eig_gemm_kernel<<<ggrid, 256, 0, stream>>>(Abuf, Bbuf, out);
}
